// Quantizer_16999480558322
// MI455X (gfx1250) — compile-verified, matrix operand order checked by reading, not run on hardware
//
#include <hip/hip_runtime.h>

// ---------------------------------------------------------------------------
// MI455X (gfx1250) VQ-VAE forward.
// All 128-channel contractions run on v_wmma_f32_16x16x32_bf16 (wave32).
// Activations stored [B][T][128] bf16 so WMMA A/B fragments are 2x16B loads.
// Weight (A) fragments are loaded once per K-chunk and fed to 2-4 time-tile
// WMMAs; B fragments for all time-tiles are issued before the WMMA burst so
// the scheduler can use partial s_wait_dscnt and overlap LDS with matrix ops.
// ---------------------------------------------------------------------------

typedef __attribute__((ext_vector_type(16))) __bf16 v16bf;
typedef __attribute__((ext_vector_type(8)))  __bf16 v8bf;
typedef __attribute__((ext_vector_type(8)))  float  v8f;

__device__ __forceinline__ v16bf load_frag16(const __bf16* p0, const __bf16* p1) {
  // ISA 16-bit A/B layout: per lane, elements 0..7 hold K=kb..kb+7,
  // elements 8..15 hold K=kb+16..kb+23  (kb = (lane/16)*8).
  v8bf lo = *(const v8bf*)p0;
  v8bf hi = *(const v8bf*)p1;
  v16bf r;
#pragma unroll
  for (int i = 0; i < 8; ++i) { r[i] = lo[i]; r[i + 8] = hi[i]; }
  return r;
}

__device__ __forceinline__ v8f wmma_bf16(v16bf a, v16bf b, v8f c) {
  return __builtin_amdgcn_wmma_f32_16x16x32_bf16(false, a, false, b,
                                                 (short)0, c, false, false);
}

__device__ __forceinline__ v8bf zero8bf() {
  v8bf v;
#pragma unroll
  for (int i = 0; i < 8; ++i) v[i] = (__bf16)0.0f;
  return v;
}

__device__ __forceinline__ v8bf relu_pack(v8f acc, bool relu) {
  v8bf ov;
#pragma unroll
  for (int rj = 0; rj < 8; ++rj) {
    float v = acc[rj];
    if (relu) v = v > 0.f ? v : 0.f;
    ov[rj] = (__bf16)v;
  }
  return ov;
}

// ---------------------------------------------------------------------------
// Encoder first conv: C_in = 1, k=4, s=2, p=1, ReLU. f32 in -> bf16 [B][T/2][128].
// ---------------------------------------------------------------------------
__global__ __launch_bounds__(128)
void enc0_kernel(const float* __restrict__ f0, const float* __restrict__ w0,
                 __bf16* __restrict__ a0, int T) {
  const int b = blockIdx.y, t = blockIdx.x, o = threadIdx.x;
  const float* xb = f0 + (size_t)b * T;
  float s = 0.f;
#pragma unroll
  for (int k = 0; k < 4; ++k) {
    int j = 2 * t - 1 + k;
    float xv = (j >= 0 && j < T) ? xb[j] : 0.f;
    s += w0[o * 4 + k] * xv;
  }
  s = s > 0.f ? s : 0.f;
  a0[((size_t)b * (T / 2) + t) * 128 + o] = (__bf16)s;
}

// ---------------------------------------------------------------------------
// 128->128 conv, k=4, stride 2, pad 1, ReLU. 8 waves, 64 outputs/block.
// wt layout: [k][o][c] bf16. Each A fragment feeds 4 time-tile WMMAs.
// ---------------------------------------------------------------------------
__global__ __launch_bounds__(256)
void conv_s2_kernel(const __bf16* __restrict__ x, const __bf16* __restrict__ wt,
                    __bf16* __restrict__ y, int Tin, int Tout) {
  __shared__ __bf16 tile[130 * 128];
  const int b = blockIdx.y;
  const int t0 = blockIdx.x * 64;
  const int tid = threadIdx.x;
  const __bf16* xb = x + (size_t)b * Tin * 128;

  // Stage input rows j = 2*t0-1 .. 2*t0+128 (130 rows), zero OOB.
  for (int q = tid; q < 130 * 16; q += 256) {
    int r = q >> 4, c8 = (q & 15) << 3;
    int j = 2 * t0 - 1 + r;
    v8bf v = (j >= 0 && j < Tin) ? *(const v8bf*)(xb + (size_t)j * 128 + c8)
                                 : zero8bf();
    *(v8bf*)(&tile[r * 128 + c8]) = v;
  }
  __syncthreads();

  const int wave = tid >> 5, lane = tid & 31;
  const int o0 = wave * 16;
  const int nn = lane & 15;          // A: M row / B: N col for this lane
  const int kb = (lane >> 4) * 8;    // K sub-group offset
  const int mb = kb;                 // D: row half offset
  __bf16* yb = y + (size_t)b * Tout * 128;

  v8f acc[4] = {};
#pragma unroll
  for (int k = 0; k < 4; ++k) {
#pragma unroll
    for (int cc = 0; cc < 4; ++cc) {
      const int c0 = cc * 32;
      const __bf16* ap = wt + ((size_t)(k * 128 + o0 + nn) * 128 + c0 + kb);
      v16bf A = load_frag16(ap, ap + 16);   // one weight fetch ...
      v16bf Bv[4];
#pragma unroll
      for (int tt = 0; tt < 4; ++tt) {      // issue all B loads first
        const int r = 2 * (16 * tt + nn) + k;
        const __bf16* bp = &tile[r * 128 + c0 + kb];
        Bv[tt] = load_frag16(bp, bp + 16);
      }
#pragma unroll
      for (int tt = 0; tt < 4; ++tt)        // ... then the WMMA burst
        acc[tt] = wmma_bf16(A, Bv[tt], acc[tt]);
    }
  }
#pragma unroll
  for (int tt = 0; tt < 4; ++tt) {
    const int t = t0 + 16 * tt + nn;
    *(v8bf*)(yb + (size_t)t * 128 + o0 + mb) = relu_pack(acc[tt], true);
  }
}

// ---------------------------------------------------------------------------
// Transposed conv (k=4, s=2, p=1) == conv with lhs_dilation=2, pad=2. ReLU.
// Even outputs t=2s use taps {0,2} on x[s-1],x[s]; odd t=2s+1 taps {1,3} on
// x[s],x[s+1]. Block covers 32 input positions -> 64 outputs.
// ---------------------------------------------------------------------------
__global__ __launch_bounds__(256)
void tconv_kernel(const __bf16* __restrict__ x, const __bf16* __restrict__ wt,
                  __bf16* __restrict__ y, int Tin) {
  __shared__ __bf16 tile[34 * 128];
  const int b = blockIdx.y;
  const int s0 = blockIdx.x * 32;
  const int tid = threadIdx.x;
  const __bf16* xb = x + (size_t)b * Tin * 128;

  for (int q = tid; q < 34 * 16; q += 256) {
    int r = q >> 4, c8 = (q & 15) << 3;
    int j = s0 - 1 + r;
    v8bf v = (j >= 0 && j < Tin) ? *(const v8bf*)(xb + (size_t)j * 128 + c8)
                                 : zero8bf();
    *(v8bf*)(&tile[r * 128 + c8]) = v;
  }
  __syncthreads();

  const int wave = tid >> 5, lane = tid & 31;
  const int o0 = wave * 16, nn = lane & 15;
  const int kb = (lane >> 4) * 8, mb = kb;
  const int Tout = 2 * Tin;
  __bf16* yb = y + (size_t)b * Tout * 128;

#pragma unroll
  for (int par = 0; par < 2; ++par) {
    v8f acc[2] = {};
#pragma unroll
    for (int tap = 0; tap < 2; ++tap) {
      const int k   = par + 2 * tap;                      // {0,2} or {1,3}
      const int off = (par == 0) ? tap : tap + 1;         // local row offset
#pragma unroll
      for (int cc = 0; cc < 4; ++cc) {
        const int c0 = cc * 32;
        const __bf16* ap = wt + ((size_t)(k * 128 + o0 + nn) * 128 + c0 + kb);
        v16bf A = load_frag16(ap, ap + 16);   // one weight fetch ...
        v16bf Bv[2];
#pragma unroll
        for (int ss = 0; ss < 2; ++ss) {      // issue both B loads first
          const int r = 16 * ss + nn + off;
          const __bf16* bp = &tile[r * 128 + c0 + kb];
          Bv[ss] = load_frag16(bp, bp + 16);
        }
#pragma unroll
        for (int ss = 0; ss < 2; ++ss)
          acc[ss] = wmma_bf16(A, Bv[ss], acc[ss]);
      }
    }
#pragma unroll
    for (int ss = 0; ss < 2; ++ss) {
      const int t = 2 * (s0 + 16 * ss + nn) + par;
      *(v8bf*)(yb + (size_t)t * 128 + o0 + mb) = relu_pack(acc[ss], true);
    }
  }
}

// ---------------------------------------------------------------------------
// 1x1 projection: out[row,o] = sum_c w[o,c]*x[row,c]. Optional f32 side output.
// 64 rows/block; each A fragment feeds 4 WMMAs.
// ---------------------------------------------------------------------------
__global__ __launch_bounds__(256)
void gemm1x1_kernel(const __bf16* __restrict__ x, const __bf16* __restrict__ w,
                    __bf16* __restrict__ ybf, float* __restrict__ yf32) {
  __shared__ __bf16 tile[64 * 128];
  const int r0 = blockIdx.x * 64;
  const int tid = threadIdx.x;
  for (int q = tid; q < 64 * 16; q += 256) {
    int r = q >> 4, c8 = (q & 15) << 3;
    *(v8bf*)(&tile[r * 128 + c8]) = *(const v8bf*)(x + (size_t)(r0 + r) * 128 + c8);
  }
  __syncthreads();

  const int wave = tid >> 5, lane = tid & 31;
  const int o0 = wave * 16, nn = lane & 15;
  const int kb = (lane >> 4) * 8, mb = kb;

  v8f acc[4] = {};
#pragma unroll
  for (int cc = 0; cc < 4; ++cc) {
    const int c0 = cc * 32;
    const __bf16* ap = w + ((size_t)(o0 + nn) * 128 + c0 + kb);
    v16bf A = load_frag16(ap, ap + 16);
    v16bf Bv[4];
#pragma unroll
    for (int tt = 0; tt < 4; ++tt) {
      const __bf16* bp = &tile[(16 * tt + nn) * 128 + c0 + kb];
      Bv[tt] = load_frag16(bp, bp + 16);
    }
#pragma unroll
    for (int tt = 0; tt < 4; ++tt)
      acc[tt] = wmma_bf16(A, Bv[tt], acc[tt]);
  }
#pragma unroll
  for (int tt = 0; tt < 4; ++tt) {
    const size_t row = (size_t)r0 + 16 * tt + nn;
    *(v8bf*)(ybf + row * 128 + o0 + mb) = relu_pack(acc[tt], false);
    if (yf32) {
      float* fp = yf32 + row * 128 + o0 + mb;
#pragma unroll
      for (int rj = 0; rj < 8; ++rj) fp[rj] = acc[tt][rj];
    }
  }
}

// ---------------------------------------------------------------------------
// VQ argmin: score(code) = ||c||^2 - 2 z.c  (||z||^2 constant per row).
// A = codebook tiles (M=code), B = 16 z-rows held in registers (N=row).
// Lane halves cover code%16<8 / >=8; merged with shfl_xor(16).
// ---------------------------------------------------------------------------
__global__ __launch_bounds__(256)
void vq_argmin_kernel(const __bf16* __restrict__ zbf, const __bf16* __restrict__ cb,
                      const float* __restrict__ cnorm, int* __restrict__ code,
                      float* __restrict__ usage) {
  const int wave = threadIdx.x >> 5, lane = threadIdx.x & 31;
  const int row0 = blockIdx.x * 128 + wave * 16;
  const int nn = lane & 15, kb = (lane >> 4) * 8, mb = kb;

  v16bf Bf[4];
#pragma unroll
  for (int cc = 0; cc < 4; ++cc) {
    const __bf16* bp = zbf + (size_t)(row0 + nn) * 128 + cc * 32 + kb;
    Bf[cc] = load_frag16(bp, bp + 16);
  }

  float best = 3.4e38f;
  int bestIdx = 0;
  for (int m0 = 0; m0 < 512; m0 += 16) {
    v16bf Af[4];
#pragma unroll
    for (int cc = 0; cc < 4; ++cc) {     // issue all codebook loads first
      const __bf16* ap = cb + (size_t)(m0 + nn) * 128 + cc * 32 + kb;
      Af[cc] = load_frag16(ap, ap + 16);
    }
    v8f acc = {};
#pragma unroll
    for (int cc = 0; cc < 4; ++cc) acc = wmma_bf16(Af[cc], Bf[cc], acc);
#pragma unroll
    for (int rj = 0; rj < 8; ++rj) {
      const int ci = m0 + mb + rj;
      const float s = cnorm[ci] - 2.0f * acc[rj];
      if (s < best) { best = s; bestIdx = ci; }
    }
  }
  const float ob = __shfl_xor(best, 16, 32);
  const int   oi = __shfl_xor(bestIdx, 16, 32);
  if (ob < best || (ob == best && oi < bestIdx)) { best = ob; bestIdx = oi; }
  if (lane < 16) {
    code[row0 + nn] = bestIdx;
    usage[bestIdx] = 1.0f;   // benign race: everyone writes 1.0
  }
}

// ---------------------------------------------------------------------------
// Gather zq (f32 codebook -> bf16 h) and per-row commit partial sums.
// ---------------------------------------------------------------------------
__global__ __launch_bounds__(128)
void gather_commit_kernel(const float* __restrict__ zf32, const float* __restrict__ cb,
                          const int* __restrict__ code, __bf16* __restrict__ h,
                          float* __restrict__ partials) {
  const int row = blockIdx.x, c = threadIdx.x;
  const int k = code[row];
  const float zv = zf32[(size_t)row * 128 + c];
  const float cv = cb[(size_t)k * 128 + c];
  h[(size_t)row * 128 + c] = (__bf16)cv;
  float d = zv - cv;
  d *= d;
  for (int off = 16; off > 0; off >>= 1) d += __shfl_down(d, off, 32);
  __shared__ float red[4];
  if ((c & 31) == 0) red[c >> 5] = d;
  __syncthreads();
  if (c == 0) partials[row] = red[0] + red[1] + red[2] + red[3];
}

// ---------------------------------------------------------------------------
// Final output conv: [1,128,3], pad 1, no ReLU. f32 out.
// ---------------------------------------------------------------------------
__global__ __launch_bounds__(256)
void out_conv_kernel(const __bf16* __restrict__ x, const float* __restrict__ w,
                     float* __restrict__ out, int T) {
  const int idx = blockIdx.x * 256 + threadIdx.x;  // b*T + t
  const int b = idx / T, t = idx - b * T;
  const __bf16* xb = x + (size_t)b * T * 128;
  float s = 0.f;
#pragma unroll
  for (int k = 0; k < 3; ++k) {
    const int j = t + k - 1;
    if ((unsigned)j >= (unsigned)T) continue;
    const __bf16* xr = xb + (size_t)j * 128;
#pragma unroll 4
    for (int c8 = 0; c8 < 128; c8 += 8) {
      v8bf v = *(const v8bf*)(xr + c8);
#pragma unroll
      for (int i = 0; i < 8; ++i) s += w[(c8 + i) * 3 + k] * (float)v[i];
    }
  }
  out[idx] = s;
}

// ---------------------------------------------------------------------------
// Weight prep: conv [O=128][C=128][K=4] f32 -> [k][o][c] bf16.
// ---------------------------------------------------------------------------
__global__ __launch_bounds__(256)
void prep_convw_kernel(const float* __restrict__ w, __bf16* __restrict__ wt) {
  const int idx = blockIdx.x * 256 + threadIdx.x;  // 65536
  const int o = idx >> 9, c = (idx >> 2) & 127, k = idx & 3;
  wt[((size_t)k * 128 + o) * 128 + c] = (__bf16)w[idx];
}

__global__ __launch_bounds__(256)
void prep_proj_kernel(const float* __restrict__ w, __bf16* __restrict__ wt) {
  const int idx = blockIdx.x * 256 + threadIdx.x;  // 16384
  wt[idx] = (__bf16)w[idx];
}

__global__ __launch_bounds__(128)
void prep_cb_kernel(const float* __restrict__ cb, __bf16* __restrict__ cbbf,
                    float* __restrict__ cnorm) {
  const int k = blockIdx.x, c = threadIdx.x;
  const float v = cb[(size_t)k * 128 + c];
  cbbf[(size_t)k * 128 + c] = (__bf16)v;
  float d = v * v;
  for (int off = 16; off > 0; off >>= 1) d += __shfl_down(d, off, 32);
  __shared__ float red[4];
  if ((c & 31) == 0) red[c >> 5] = d;
  __syncthreads();
  if (c == 0) cnorm[k] = red[0] + red[1] + red[2] + red[3];
}

// ---------------------------------------------------------------------------
// Deterministic scalar finalize: commit = sum(partials)/(32768*128),
// usage = sum(flags)/512.
// ---------------------------------------------------------------------------
__global__ __launch_bounds__(256)
void finalize_kernel(const float* __restrict__ partials, const float* __restrict__ usage,
                     float* __restrict__ out2) {
  __shared__ float red[256];
  float s = 0.f;
  for (int i = threadIdx.x; i < 32768; i += 256) s += partials[i];
  red[threadIdx.x] = s;
  __syncthreads();
  for (int off = 128; off > 0; off >>= 1) {
    if ((int)threadIdx.x < off) red[threadIdx.x] += red[threadIdx.x + off];
    __syncthreads();
  }
  if (threadIdx.x == 0) {
    out2[0] = red[0] / (32768.0f * 128.0f);
    float u = 0.f;
    for (int i = 0; i < 512; ++i) u += usage[i];
    out2[1] = u / 512.0f;
  }
}

// ---------------------------------------------------------------------------
extern "C" void kernel_launch(void* const* d_in, const int* in_sizes, int n_in,
                              void* d_out, int out_size, void* d_ws, size_t ws_size,
                              hipStream_t stream) {
  (void)in_sizes; (void)n_in; (void)out_size; (void)ws_size;
  const float* f0       = (const float*)d_in[0];
  const float* enc_w0   = (const float*)d_in[1];
  const float* enc_w1   = (const float*)d_in[2];
  const float* enc_w2   = (const float*)d_in[3];
  const float* enc_w3   = (const float*)d_in[4];
  const float* enc_proj = (const float*)d_in[5];
  const float* codebook = (const float*)d_in[6];
  const float* dec_proj = (const float*)d_in[7];
  const float* dec_w0   = (const float*)d_in[8];
  const float* dec_w1   = (const float*)d_in[9];
  const float* dec_w2   = (const float*)d_in[10];
  const float* dec_w3   = (const float*)d_in[11];
  const float* dec_out  = (const float*)d_in[12];
  float* out = (float*)d_out;

  char* ws = (char*)d_ws;
  const size_t MB = 1ull << 20;
  // Activation arena (regions reused once their producers/consumers retire):
  __bf16* a0  = (__bf16*)(ws + 0);          // [16,16384,128] 64MB
  __bf16* a1  = (__bf16*)(ws + 64 * MB);    // 32MB
  __bf16* a2  = (__bf16*)(ws + 96 * MB);    // 16MB
  __bf16* a3  = (__bf16*)(ws + 112 * MB);   // 8MB
  float*  zf  = (float*) (ws + 120 * MB);   // z f32, 16MB
  __bf16* zb  = (__bf16*)(ws + 136 * MB);   // z bf16, 8MB
  __bf16* h   = (__bf16*)(ws + 144 * MB);   // zq bf16, 8MB
  __bf16* dd0 = (__bf16*)(ws + 152 * MB);   // 8MB
  __bf16* dd1 = (__bf16*)(ws + 0);          // 16MB (a0 dead)
  __bf16* dd2 = (__bf16*)(ws + 16 * MB);    // 32MB (a0 dead)
  __bf16* dd3 = (__bf16*)(ws + 64 * MB);    // 64MB (a1..a3, zf dead)
  __bf16* dd4 = (__bf16*)(ws + 160 * MB);   // 128MB
  char* WB = ws + 288 * MB;                 // weights + small buffers
  __bf16* wtE1 = (__bf16*)(WB + 0 * 131072);
  __bf16* wtE2 = (__bf16*)(WB + 1 * 131072);
  __bf16* wtE3 = (__bf16*)(WB + 2 * 131072);
  __bf16* wtD0 = (__bf16*)(WB + 3 * 131072);
  __bf16* wtD1 = (__bf16*)(WB + 4 * 131072);
  __bf16* wtD2 = (__bf16*)(WB + 5 * 131072);
  __bf16* wtD3 = (__bf16*)(WB + 6 * 131072);
  __bf16* wpE  = (__bf16*)(WB + 917504);
  __bf16* wpD  = (__bf16*)(WB + 950272);
  __bf16* cbbf = (__bf16*)(WB + 983040);
  float*  cnorm = (float*)(WB + 1114112);
  int*    code  = (int*)  (WB + 1116160);
  float*  usage = (float*)(WB + 1247232);
  float*  partials = (float*)(WB + 1249280);

  // ---- prep ----
  prep_convw_kernel<<<256, 256, 0, stream>>>(enc_w1, wtE1);
  prep_convw_kernel<<<256, 256, 0, stream>>>(enc_w2, wtE2);
  prep_convw_kernel<<<256, 256, 0, stream>>>(enc_w3, wtE3);
  prep_convw_kernel<<<256, 256, 0, stream>>>(dec_w0, wtD0);
  prep_convw_kernel<<<256, 256, 0, stream>>>(dec_w1, wtD1);
  prep_convw_kernel<<<256, 256, 0, stream>>>(dec_w2, wtD2);
  prep_convw_kernel<<<256, 256, 0, stream>>>(dec_w3, wtD3);
  prep_proj_kernel<<<64, 256, 0, stream>>>(enc_proj, wpE);
  prep_proj_kernel<<<64, 256, 0, stream>>>(dec_proj, wpD);
  prep_cb_kernel<<<512, 128, 0, stream>>>(codebook, cbbf, cnorm);
  hipMemsetAsync(usage, 0, 512 * sizeof(float), stream);

  // ---- encoder ----
  enc0_kernel<<<dim3(16384, 16), 128, 0, stream>>>(f0, enc_w0, a0, 32768);
  conv_s2_kernel<<<dim3(128, 16), 256, 0, stream>>>(a0, wtE1, a1, 16384, 8192);
  conv_s2_kernel<<<dim3(64, 16), 256, 0, stream>>>(a1, wtE2, a2, 8192, 4096);
  conv_s2_kernel<<<dim3(32, 16), 256, 0, stream>>>(a2, wtE3, a3, 4096, 2048);
  gemm1x1_kernel<<<512, 256, 0, stream>>>(a3, wpE, zb, zf);   // 32768 rows

  // ---- VQ bottleneck ----
  vq_argmin_kernel<<<256, 256, 0, stream>>>(zb, cbbf, cnorm, code, usage);
  gather_commit_kernel<<<32768, 128, 0, stream>>>(zf, codebook, code, h, partials);

  // ---- decoder ----
  gemm1x1_kernel<<<512, 256, 0, stream>>>(h, wpD, dd0, nullptr);
  tconv_kernel<<<dim3(64, 16), 256, 0, stream>>>(dd0, wtD0, dd1, 2048);
  tconv_kernel<<<dim3(128, 16), 256, 0, stream>>>(dd1, wtD1, dd2, 4096);
  tconv_kernel<<<dim3(256, 16), 256, 0, stream>>>(dd2, wtD2, dd3, 8192);
  tconv_kernel<<<dim3(512, 16), 256, 0, stream>>>(dd3, wtD3, dd4, 16384);
  out_conv_kernel<<<2048, 256, 0, stream>>>(dd4, dec_out, out, 32768);

  // ---- scalars (out, commit, usage) ----
  finalize_kernel<<<1, 256, 0, stream>>>(partials, usage, out + 524288);
}
